// GCN_63161789055384
// MI455X (gfx1250) — matrix operand order repaired and checked by reading, not compiled, not run on hardware
//
#include <hip/hip_runtime.h>
#include <hip/hip_bf16.h>

typedef __attribute__((ext_vector_type(16))) _Float16 v16h;
typedef __attribute__((ext_vector_type(8)))  float    v8f;

#define N_NODES 50000
#define N_EDGES 800000
#define FDIM 128
#define NTILES (N_NODES / 16)   // 3125, exact

// ---------------------------------------------------------------- utilities
__global__ void gcn_zero_f32(float* __restrict__ p, int n) {
    int i = blockIdx.x * blockDim.x + threadIdx.x;
    if (i < n) p[i] = 0.0f;
}

// degree count via atomics: dsrc[src[e]] += 1, ddst[dst[e]] += 1
__global__ void gcn_degree(const int* __restrict__ src, const int* __restrict__ dst,
                           float* __restrict__ dsrc, float* __restrict__ ddst, int ne) {
    int i = blockIdx.x * blockDim.x + threadIdx.x;
    if (i < ne) {
        atomicAdd(&dsrc[src[i]], 1.0f);
        atomicAdd(&ddst[dst[i]], 1.0f);
    }
}

// in-place deg -> rsqrt(max(deg, 1))
__global__ void gcn_norm(float* __restrict__ p, int n) {
    int i = blockIdx.x * blockDim.x + threadIdx.x;
    if (i < n) p[i] = rsqrtf(fmaxf(p[i], 1.0f));
}

// ---------------------------------------------------------------- edge scatter
// msg[dst[e], f] += x[src[e], f] * outnorm[src[e]]
// blockDim.x == 128 (= feature dim): coalesced row reads, scalarized indices.
__global__ void gcn_edge_scatter(const int* __restrict__ src, const int* __restrict__ dst,
                                 const float* __restrict__ x, const float* __restrict__ outnorm,
                                 float* __restrict__ msg, int ne) {
    int f = threadIdx.x;
    for (int e = blockIdx.x; e < ne; e += gridDim.x) {
        int s = src[e];
        int d = dst[e];
        float c = outnorm[s];
        atomicAdd(&msg[(size_t)d * FDIM + f], x[(size_t)s * FDIM + f] * c);
    }
}

// ---------------------------------------------------------------- fused WMMA GEMM
// out[i,:] = act( (msg[i,:] * innorm[i]) @ W + b )
// One wave -> one 16-row tile x all 128 cols (8 accumulators of 16x16 f32).
//
// W (128x128 f32, row-major [k][n]) is staged once per block into LDS,
// PRE-SWIZZLED into the exact per-lane WMMA B-fragment layout:
//   Wswz[(ks*8 + nt)*32 + lane] is the v16h this lane feeds to WMMA for
//   K-step ks, column-tile nt, where half h of that vector is
//   W[k][n] with k = ks*32 + 16*(h/8) + 8*(lane/16) + (h%8), n = nt*16 + (lane%16).
// Each fragment read is then a single contiguous 32-byte LDS load per lane.
__global__ void __launch_bounds__(256)
gcn_gemm_wmma(const float* __restrict__ msg, const float* __restrict__ innorm,
              const float* __restrict__ W, const float* __restrict__ bias,
              float* __restrict__ out, int ntiles, int relu) {
    __shared__ v16h Wswz[4 * 8 * 32];          // 32 KB of 320 KB WGP LDS

    // cooperative f32 -> f16 staging, swizzled to fragment layout
    {
        _Float16* Wl = (_Float16*)Wswz;
        for (int idx = threadIdx.x; idx < 4 * 8 * 32 * 16; idx += blockDim.x) {
            int h  = idx & 15;
            int ln = (idx >> 4) & 31;
            int nt = (idx >> 9) & 7;
            int ks = idx >> 12;
            int k  = ks * 32 + ((h >> 3) << 4) + ((ln >> 4) << 3) + (h & 7);
            int n  = nt * 16 + (ln & 15);
            Wl[idx] = (_Float16)W[k * FDIM + n];
        }
    }
    __syncthreads();

    const int wave = blockIdx.x * (blockDim.x >> 5) + (threadIdx.x >> 5);
    const int lane = threadIdx.x & 31;
    if (wave >= ntiles) return;                 // uniform per wave: EXEC all-ones inside

    const int l16  = lane & 15;
    const int lhi  = lane >> 4;                 // 0 or 1
    const int row0 = wave * 16;
    const int arow = row0 + l16;                // A-matrix row held by this lane
    const float nrm = innorm[arow];
    const float4* mrow = (const float4*)(msg + (size_t)arow * FDIM);

    v8f acc[8];
    const v8f zero = {};
    #pragma unroll
    for (int t = 0; t < 8; ++t) acc[t] = zero;

    #pragma unroll
    for (int ks = 0; ks < 4; ++ks) {            // K = 128 in 4 steps of 32
        // A fragment (16-bit A 16x32 layout): half h -> k = 16*(h/8) + 8*lhi + (h%8)
        // -> two contiguous 8-float runs: [ks*32 + 8*lhi, +8) and [ks*32 + 16 + 8*lhi, +8)
        float4 p0 = mrow[ks * 8 + lhi * 2 + 0];
        float4 p1 = mrow[ks * 8 + lhi * 2 + 1];
        float4 p2 = mrow[ks * 8 + 4 + lhi * 2 + 0];
        float4 p3 = mrow[ks * 8 + 4 + lhi * 2 + 1];
        v16h a;
        a[0]  = (_Float16)(p0.x * nrm);  a[1]  = (_Float16)(p0.y * nrm);
        a[2]  = (_Float16)(p0.z * nrm);  a[3]  = (_Float16)(p0.w * nrm);
        a[4]  = (_Float16)(p1.x * nrm);  a[5]  = (_Float16)(p1.y * nrm);
        a[6]  = (_Float16)(p1.z * nrm);  a[7]  = (_Float16)(p1.w * nrm);
        a[8]  = (_Float16)(p2.x * nrm);  a[9]  = (_Float16)(p2.y * nrm);
        a[10] = (_Float16)(p2.z * nrm);  a[11] = (_Float16)(p2.w * nrm);
        a[12] = (_Float16)(p3.x * nrm);  a[13] = (_Float16)(p3.y * nrm);
        a[14] = (_Float16)(p3.z * nrm);  a[15] = (_Float16)(p3.w * nrm);

        #pragma unroll
        for (int nt = 0; nt < 8; ++nt) {        // 8 column tiles of 16
            v16h b = Wswz[(ks * 8 + nt) * 32 + lane];   // one contiguous LDS frag read
            acc[nt] = __builtin_amdgcn_wmma_f32_16x16x32_f16(
                false, a, false, b, (short)0, acc[nt], false, false);
        }
    }

    // C/D layout: element r of v8f -> row = r + 8*(lane/16), col = lane%16
    #pragma unroll
    for (int nt = 0; nt < 8; ++nt) {
        int col = nt * 16 + l16;
        float bv = bias[col];
        #pragma unroll
        for (int r = 0; r < 8; ++r) {
            int row = row0 + r + (lhi << 3);
            float v = acc[nt][r] + bv;
            if (relu) v = fmaxf(v, 0.0f);
            out[(size_t)row * FDIM + col] = v;
        }
    }
}

// ---------------------------------------------------------------- column mean
// out[col] += (1/N) * sum over a strided slice of rows; out pre-zeroed.
__global__ void gcn_mean(const float* __restrict__ h, float* __restrict__ out, int nrows) {
    int col = threadIdx.x;                      // 128 threads
    float s = 0.0f;
    for (int r = blockIdx.x; r < nrows; r += gridDim.x)
        s += h[(size_t)r * FDIM + col];
    atomicAdd(&out[col], s * (1.0f / (float)nrows));
}

// ---------------------------------------------------------------- launcher
extern "C" void kernel_launch(void* const* d_in, const int* in_sizes, int n_in,
                              void* d_out, int out_size, void* d_ws, size_t ws_size,
                              hipStream_t stream) {
    const float* feat = (const float*)d_in[0];
    const float* W1   = (const float*)d_in[1];
    const float* b1   = (const float*)d_in[2];
    const float* W2   = (const float*)d_in[3];
    const float* b2   = (const float*)d_in[4];
    const float* W3   = (const float*)d_in[5];
    const float* b3   = (const float*)d_in[6];
    const int*   src  = (const int*)d_in[7];
    const int*   dst  = (const int*)d_in[8];
    float* out = (float*)d_out;

    float* ws      = (float*)d_ws;
    float* outnorm = ws;                                   // N
    float* innorm  = ws + N_NODES;                         // N
    float* msg     = ws + 2 * N_NODES;                     // N*128
    float* ha      = msg + (size_t)N_NODES * FDIM;         // N*128
    float* hb      = ha  + (size_t)N_NODES * FDIM;         // N*128

    const int nf = N_NODES * FDIM;
    dim3 z256(256);
    #define GRID(n) dim3(((n) + 255) / 256)

    // degrees -> norms (both buffers share the same clamp+rsqrt, do 2N at once)
    gcn_zero_f32<<<GRID(2 * N_NODES), z256, 0, stream>>>(outnorm, 2 * N_NODES);
    gcn_degree<<<GRID(N_EDGES), z256, 0, stream>>>(src, dst, outnorm, innorm, N_EDGES);
    gcn_norm<<<GRID(2 * N_NODES), z256, 0, stream>>>(outnorm, 2 * N_NODES);

    const dim3 egrid(2048), eblk(FDIM);
    const dim3 ggrid((NTILES + 7) / 8), gblk(256);

    // layer 1: feat -> ha (ReLU)
    gcn_zero_f32<<<GRID(nf), z256, 0, stream>>>(msg, nf);
    gcn_edge_scatter<<<egrid, eblk, 0, stream>>>(src, dst, feat, outnorm, msg, N_EDGES);
    gcn_gemm_wmma<<<ggrid, gblk, 0, stream>>>(msg, innorm, W1, b1, ha, NTILES, 1);

    // layer 2: ha -> hb (ReLU)
    gcn_zero_f32<<<GRID(nf), z256, 0, stream>>>(msg, nf);
    gcn_edge_scatter<<<egrid, eblk, 0, stream>>>(src, dst, ha, outnorm, msg, N_EDGES);
    gcn_gemm_wmma<<<ggrid, gblk, 0, stream>>>(msg, innorm, W2, b2, hb, NTILES, 1);

    // layer 3: hb -> ha (no ReLU)
    gcn_zero_f32<<<GRID(nf), z256, 0, stream>>>(msg, nf);
    gcn_edge_scatter<<<egrid, eblk, 0, stream>>>(src, dst, hb, outnorm, msg, N_EDGES);
    gcn_gemm_wmma<<<ggrid, gblk, 0, stream>>>(msg, innorm, W3, b3, ha, NTILES, 0);

    // mean over nodes -> [128]
    gcn_zero_f32<<<dim3(1), dim3(FDIM), 0, stream>>>(out, FDIM);
    gcn_mean<<<dim3(256), dim3(FDIM), 0, stream>>>(ha, out, N_NODES);
    #undef GRID
}
